// AttentiveFPGNN_74345883893926
// MI455X (gfx1250) — compile-verified, matrix operand order checked by reading, not run on hardware
//
#include <hip/hip_runtime.h>
#include <math.h>

// Problem constants (match reference setup_inputs)
#define V_NODES 100000
#define E_EDGES 400000
#define NF_DIM  74
#define EF_DIM  12
#define GF_DIM  200

// K padded to multiples of 32 (f16 WMMA k-step)
#define KP_IN   96    // pad of 74 (node) and 86 (node+edge concat)
#define KP_GF   224   // pad of 200

typedef __attribute__((ext_vector_type(8)))  _Float16 v8h;
typedef __attribute__((ext_vector_type(16))) _Float16 v16h;
typedef __attribute__((ext_vector_type(8)))  float    v8f;
typedef _Float16 half_t;
typedef long long ll;

#define NT 4   // N-tiles per wave (16x64 output per wave)

__device__ __forceinline__ float act_apply(float v, int act) {
    if (act == 1) return v > 0.f ? v : 0.01f * v;        // leaky_relu(0.01)
    if (act == 2) return v > 0.f ? v : (expf(v) - 1.f);  // elu
    return v;
}

// ---------------------------------------------------------------------------
// Pack A-side operand to f16 [M, Kpad], fusing optional row-gather, two-source
// concat, and activation. Zero-fills the K pad so the WMMA loop needs no masks.
// ---------------------------------------------------------------------------
__global__ void pack_a_f16(const float* __restrict__ A1, int K1, const int* __restrict__ idx1,
                           const float* __restrict__ A2, int K2, const int* __restrict__ idx2,
                           half_t* __restrict__ out, int M, int Kpad, int act) {
    const ll t = (ll)blockIdx.x * blockDim.x + threadIdx.x;
    if (t >= (ll)M * Kpad) return;
    const int m = (int)(t / Kpad);
    const int k = (int)(t % Kpad);
    float v = 0.f;
    if (k < K1) {
        const int r = idx1 ? idx1[m] : m;
        v = act_apply(A1[(ll)r * K1 + k], act);
    } else if (A2 && k < K1 + K2) {
        const int r = idx2 ? idx2[m] : m;
        v = act_apply(A2[(ll)r * K2 + (k - K1)], act);
    }
    out[t] = (half_t)v;
}

// Pack weight W[K,N] (f32, row-major) to transposed f16 Wt[N, Kpad], zero pad.
__global__ void pack_wt_f16(const float* __restrict__ W, half_t* __restrict__ out,
                            int K, int N, int Kpad) {
    const ll t = (ll)blockIdx.x * blockDim.x + threadIdx.x;
    if (t >= (ll)N * Kpad) return;
    const int n = (int)(t / Kpad);
    const int k = (int)(t % Kpad);
    out[t] = (half_t)(k < K ? W[(ll)k * N + n] : 0.f);
}

// ---------------------------------------------------------------------------
// Branch-free WMMA fragment loads: two contiguous 16B loads per operand.
// v16h element e maps to K = kb + (e<8?0:16) + half*8 + (e&7)  (CDNA5 ISA),
// so per lane the fragment is A[row, kb+half*8 .. +7] ++ A[row, kb+16+half*8 .. +7].
// ---------------------------------------------------------------------------
__device__ __forceinline__ v16h load_frag(const half_t* __restrict__ base, int kb) {
    v8h lo = *reinterpret_cast<const v8h*>(base + kb);
    v8h hi = *reinterpret_cast<const v8h*>(base + kb + 16);
    return __builtin_shufflevector(lo, hi, 0, 1, 2, 3, 4, 5, 6, 7,
                                           8, 9, 10, 11, 12, 13, 14, 15);
}

// Per-wave 16x(16*NT) tile compute. Row/col clamped (loads in-bounds; clamped
// lanes only feed never-stored outputs), K pre-padded with zeros.
__device__ __forceinline__
void wmma_group(const half_t* __restrict__ Ap, int Kpad, const half_t* __restrict__ Wt,
                int M, int N, int m0, int n0, int lane, v8f acc[NT]) {
    const int half = lane >> 4;
    const int rc   = lane & 15;
    int row = m0 + rc; if (row >= M) row = M - 1;
    const half_t* arow = Ap + (ll)row * Kpad + half * 8;
    const half_t* brow[NT];
#pragma unroll
    for (int j = 0; j < NT; ++j) {
        int col = n0 + j * 16 + rc; if (col >= N) col = N - 1;
        brow[j] = Wt + (ll)col * Kpad + half * 8;
    }
    for (int kb = 0; kb < Kpad; kb += 32) {
        const v16h a = load_frag(arow, kb);
#pragma unroll
        for (int j = 0; j < NT; ++j) {
            const v16h b = load_frag(brow[j], kb);
            acc[j] = __builtin_amdgcn_wmma_f32_16x16x32_f16(false, a, false, b,
                                                            (short)0, acc[j], false, false);
        }
    }
}

// ---------------------------------------------------------------------------
// GEMM:  out[M,N] = act_out( Ap@Wt^T + bias );  one wave per 16x64 tile group.
// ---------------------------------------------------------------------------
__global__ void gemm_wmma_f16(const half_t* __restrict__ Ap, int Kpad,
                              const half_t* __restrict__ Wt, const float* __restrict__ bias,
                              float* __restrict__ out, int M, int N, int act_out) {
    const int lane = threadIdx.x & 31;
    const int wave = threadIdx.x >> 5;
    const int tiles_m  = (M + 15) >> 4;
    const int groups_n = (N + NT * 16 - 1) / (NT * 16);
    const ll g = (ll)blockIdx.x * (blockDim.x >> 5) + wave;
    if (g >= (ll)tiles_m * groups_n) return;
    const int m0 = (int)(g / groups_n) << 4;
    const int n0 = (int)(g % groups_n) * (NT * 16);

    v8f acc[NT] = {};
    wmma_group(Ap, Kpad, Wt, M, N, m0, n0, lane, acc);

    const int half = lane >> 4;
    const int rc   = lane & 15;
#pragma unroll
    for (int j = 0; j < NT; ++j) {
        const int n = n0 + j * 16 + rc;
        if (n < N) {
            const float bv = bias ? bias[n] : 0.f;
#pragma unroll
            for (int i = 0; i < 8; ++i) {
                const int m = m0 + i + half * 8;
                if (m < M) out[(ll)m * N + n] = act_apply(acc[j][i] + bv, act_out);
            }
        }
    }
}

// Fused: tile of (he1 @ et_w + et_b), scaled by softmax wexp[m]/ssum[dst[m]],
// scatter-added into c[dst[m], n].  (AttentiveGRU1 message aggregation.)
__global__ void gemm_wmma_attn_scatter(const half_t* __restrict__ Ap, int Kpad,
                                       const half_t* __restrict__ Wt, const float* __restrict__ bias,
                                       const float* __restrict__ wexp, const float* __restrict__ ssum,
                                       const int* __restrict__ dst,
                                       float* __restrict__ cout, int M, int N) {
    const int lane = threadIdx.x & 31;
    const int wave = threadIdx.x >> 5;
    const int tiles_m  = (M + 15) >> 4;
    const int groups_n = (N + NT * 16 - 1) / (NT * 16);
    const ll g = (ll)blockIdx.x * (blockDim.x >> 5) + wave;
    if (g >= (ll)tiles_m * groups_n) return;
    const int m0 = (int)(g / groups_n) << 4;
    const int n0 = (int)(g % groups_n) * (NT * 16);

    v8f acc[NT] = {};
    wmma_group(Ap, Kpad, Wt, M, N, m0, n0, lane, acc);

    const int half = lane >> 4;
    const int rc   = lane & 15;
#pragma unroll
    for (int j = 0; j < NT; ++j) {
        const int n = n0 + j * 16 + rc;
        if (n < N) {
            const float bv = bias[n];
#pragma unroll
            for (int i = 0; i < 8; ++i) {
                const int m = m0 + i + half * 8;
                if (m < M) {
                    const int d = dst[m];
                    const float sc = wexp[m] / ssum[d];
                    atomicAdd(&cout[(ll)d * N + n], sc * (acc[j][i] + bv));
                }
            }
        }
    }
}

// ---------------------------------------------------------------------------
// Edge attention logits: leaky( Xa[ia[e]] . w[0:K] + Xb[ib[e]] . w[K:2K] + b )
// ---------------------------------------------------------------------------
__global__ void edge_logits_kernel(const float* __restrict__ Xa, const int* __restrict__ ia,
                                   const float* __restrict__ Xb, const int* __restrict__ ib,
                                   const float* __restrict__ w, const float* __restrict__ bias,
                                   float* __restrict__ out, int E, int K) {
    const int e = blockIdx.x * blockDim.x + threadIdx.x;
    if (e >= E) return;
    const float* xa = Xa + (ll)(ia ? ia[e] : e) * K;
    const float* xb = Xb + (ll)(ib ? ib[e] : e) * K;
    float acc = bias[0];
    for (int k = 0; k < K; ++k) acc += xa[k] * w[k] + xb[k] * w[K + k];
    out[e] = acc > 0.f ? acc : 0.01f * acc;
}

// ------------------- edge softmax (segment max + exp-sum) -------------------
__device__ __forceinline__ unsigned fenc(float f) {
    unsigned u = __float_as_uint(f);
    return (u & 0x80000000u) ? ~u : (u | 0x80000000u);
}
__device__ __forceinline__ float fdec(unsigned u) {
    unsigned b = (u & 0x80000000u) ? (u ^ 0x80000000u) : ~u;
    return __uint_as_float(b);
}

__global__ void seg_max_kernel(const float* __restrict__ logits, const int* __restrict__ dst,
                               unsigned* __restrict__ menc, int E) {
    const int e = blockIdx.x * blockDim.x + threadIdx.x;
    if (e >= E) return;
    atomicMax(&menc[dst[e]], fenc(logits[e]));
}

__global__ void exp_sum_kernel(const float* __restrict__ logits, const int* __restrict__ dst,
                               const unsigned* __restrict__ menc,
                               float* __restrict__ wexp, float* __restrict__ ssum, int E) {
    const int e = blockIdx.x * blockDim.x + threadIdx.x;
    if (e >= E) return;
    const int d = dst[e];
    const float w = expf(logits[e] - fdec(menc[d]));
    wexp[e] = w;
    atomicAdd(&ssum[d], w);
}

// Layer-2 message aggregation: c2[dst[e]] += (wexp/ssum) * hvproj[src[e]]
__global__ void attn_gather_scatter(const float* __restrict__ hvproj,
                                    const float* __restrict__ wexp, const float* __restrict__ ssum,
                                    const int* __restrict__ src, const int* __restrict__ dst,
                                    float* __restrict__ c2, int E, int N) {
    const ll t = (ll)blockIdx.x * blockDim.x + threadIdx.x;
    if (t >= (ll)E * N) return;
    const int e = (int)(t / N);
    const int f = (int)(t % N);
    const float sc = wexp[e] / ssum[dst[e]];
    atomicAdd(&c2[(ll)dst[e] * N + f], sc * hvproj[(ll)src[e] * N + f]);
}

// GRU gate combine: h = relu((1-z)*n + z*hprev), gates from gi/gh [V,3F]
__global__ void gru_combine_kernel(const float* __restrict__ gi, const float* __restrict__ gh,
                                   const float* __restrict__ hprev, float* __restrict__ hout,
                                   int Vn, int F) {
    const ll t = (ll)blockIdx.x * blockDim.x + threadIdx.x;
    if (t >= (ll)Vn * F) return;
    const int v = (int)(t / F);
    const int f = (int)(t % F);
    const float* giv = gi + (ll)v * 3 * F;
    const float* ghv = gh + (ll)v * 3 * F;
    const float r = 1.f / (1.f + expf(-(giv[f]     + ghv[f])));
    const float z = 1.f / (1.f + expf(-(giv[F + f] + ghv[F + f])));
    const float n = tanhf(giv[2 * F + f] + r * ghv[2 * F + f]);
    const float h = (1.f - z) * n + z * hprev[t];
    hout[t] = h > 0.f ? h : 0.f;
}

// ---------------------------------------------------------------------------
extern "C" void kernel_launch(void* const* d_in, const int* in_sizes, int n_in,
                              void* d_out, int out_size, void* d_ws, size_t ws_size,
                              hipStream_t stream) {
    (void)in_sizes; (void)n_in; (void)out_size; (void)ws_size;
    const float* node_feats = (const float*)d_in[0];
    const float* edge_feats = (const float*)d_in[1];
    const float* pn_w  = (const float*)d_in[2];  const float* pn_b  = (const float*)d_in[3];
    const float* pe1_w = (const float*)d_in[4];  const float* pe1_b = (const float*)d_in[5];
    const float* pe2_w = (const float*)d_in[6];  const float* pe2_b = (const float*)d_in[7];
    const float* et_w  = (const float*)d_in[8];  const float* et_b  = (const float*)d_in[9];
    const float* g1_wih = (const float*)d_in[10]; const float* g1_whh = (const float*)d_in[11];
    const float* g1_bih = (const float*)d_in[12]; const float* g1_bhh = (const float*)d_in[13];
    const float* lpe_w = (const float*)d_in[14]; const float* lpe_b = (const float*)d_in[15];
    const float* lpn_w = (const float*)d_in[16]; const float* lpn_b = (const float*)d_in[17];
    const float* g2_wih = (const float*)d_in[18]; const float* g2_whh = (const float*)d_in[19];
    const float* g2_bih = (const float*)d_in[20]; const float* g2_bhh = (const float*)d_in[21];
    const int* src = (const int*)d_in[22];
    const int* dst = (const int*)d_in[23];

    const int V = V_NODES, E = E_EDGES, GF = GF_DIM;

    // ---------------- workspace layout ----------------
    // f32 region
    float* ws = (float*)d_ws;
    size_t o = 0;
    float*    hv     = ws + o; o += (size_t)V * GF;        // hv_new, reused for hv_proj
    float*    he1    = ws + o; o += (size_t)E * GF;
    float*    logits = ws + o; o += (size_t)E;
    unsigned* menc   = (unsigned*)(ws + o); o += (size_t)V;
    float*    ssum   = ws + o; o += (size_t)V;
    float*    wexp   = ws + o; o += (size_t)E;
    float*    cbuf   = ws + o; o += (size_t)V * GF;        // c, then c2
    float*    gi     = ws + o; o += (size_t)V * 3 * GF;
    float*    gh     = ws + o; o += (size_t)V * 3 * GF;
    float*    hbuf   = ws + o; o += (size_t)V * GF;        // h after layer 1
    // f16 region (16B aligned: all sizes are multiples of 8 halves)
    half_t* hws = (half_t*)(ws + o);
    size_t ho = 0;
    half_t* nf_p   = hws + ho; ho += (size_t)V * KP_IN;    // node_feats packed
    half_t* ce_p   = hws + ho; ho += (size_t)E * KP_IN;    // concat(node[src], edge) packed
    half_t* he1_p  = hws + ho; ho += (size_t)E * KP_GF;
    half_t* hv_p   = hws + ho; ho += (size_t)V * KP_GF;    // hv_new / h packed (per layer)
    half_t* cx_p   = hws + ho; ho += (size_t)V * KP_GF;    // elu(c) packed
    half_t* pn_wt  = hws + ho; ho += (size_t)GF * KP_IN;
    half_t* pe1_wt = hws + ho; ho += (size_t)GF * KP_IN;
    half_t* et_wt  = hws + ho; ho += (size_t)GF * KP_GF;
    half_t* wih_t  = hws + ho; ho += (size_t)3 * GF * KP_GF;  // per-layer reuse
    half_t* whh_t  = hws + ho; ho += (size_t)3 * GF * KP_GF;
    half_t* lpn_wt = hws + ho; ho += (size_t)GF * KP_GF;

    const int TB = 256;  // 8 waves
    auto ew_blocks = [](ll n) { return (int)((n + 255) / 256); };
    auto gemm_blocks = [](int M, int N) {
        ll groups = (ll)((M + 15) / 16) * ((N + NT * 16 - 1) / (NT * 16));
        return (int)((groups + 7) / 8);
    };

    // ---------------- weight packs (tiny) ----------------
    pack_wt_f16<<<ew_blocks((ll)GF * KP_IN), TB, 0, stream>>>(pn_w,  pn_wt,  NF_DIM, GF, KP_IN);
    pack_wt_f16<<<ew_blocks((ll)GF * KP_IN), TB, 0, stream>>>(pe1_w, pe1_wt, NF_DIM + EF_DIM, GF, KP_IN);
    pack_wt_f16<<<ew_blocks((ll)GF * KP_GF), TB, 0, stream>>>(et_w,  et_wt,  GF, GF, KP_GF);
    pack_wt_f16<<<ew_blocks((ll)GF * KP_GF), TB, 0, stream>>>(lpn_w, lpn_wt, GF, GF, KP_GF);

    // ---------------- Layer 1: GetContext + AttentiveGRU1 ----------------
    // hv = leaky(node_feats @ pn_w + pn_b)
    pack_a_f16<<<ew_blocks((ll)V * KP_IN), TB, 0, stream>>>(
        node_feats, NF_DIM, nullptr, nullptr, 0, nullptr, nf_p, V, KP_IN, 0);
    gemm_wmma_f16<<<gemm_blocks(V, GF), TB, 0, stream>>>(nf_p, KP_IN, pn_wt, pn_b, hv, V, GF, 1);
    // he1 = leaky(concat(node_feats[src], edge_feats) @ pe1_w + pe1_b)
    pack_a_f16<<<ew_blocks((ll)E * KP_IN), TB, 0, stream>>>(
        node_feats, NF_DIM, src, edge_feats, EF_DIM, nullptr, ce_p, E, KP_IN, 0);
    gemm_wmma_f16<<<gemm_blocks(E, GF), TB, 0, stream>>>(ce_p, KP_IN, pe1_wt, pe1_b, he1, E, GF, 1);
    // logits = leaky(concat(hv[dst], he1) @ pe2_w + pe2_b); edge softmax over dst
    edge_logits_kernel<<<ew_blocks(E), TB, 0, stream>>>(hv, dst, he1, nullptr, pe2_w, pe2_b, logits, E, GF);
    hipMemsetAsync(menc, 0, (size_t)V * sizeof(unsigned), stream);
    hipMemsetAsync(ssum, 0, (size_t)V * sizeof(float), stream);
    hipMemsetAsync(cbuf, 0, (size_t)V * GF * sizeof(float), stream);
    seg_max_kernel<<<ew_blocks(E), TB, 0, stream>>>(logits, dst, menc, E);
    exp_sum_kernel<<<ew_blocks(E), TB, 0, stream>>>(logits, dst, menc, wexp, ssum, E);
    // c = segment_sum( softmax * (he1 @ et_w + et_b), dst )  [fused WMMA + scatter]
    pack_a_f16<<<ew_blocks((ll)E * KP_GF), TB, 0, stream>>>(
        he1, GF, nullptr, nullptr, 0, nullptr, he1_p, E, KP_GF, 0);
    gemm_wmma_attn_scatter<<<gemm_blocks(E, GF), TB, 0, stream>>>(
        he1_p, KP_GF, et_wt, et_b, wexp, ssum, dst, cbuf, E, GF);
    // GRU1: gi = elu(c)@wih + bih ; gh = hv@whh + bhh ; h = relu(combine)
    pack_a_f16<<<ew_blocks((ll)V * KP_GF), TB, 0, stream>>>(
        cbuf, GF, nullptr, nullptr, 0, nullptr, cx_p, V, KP_GF, 2);   // elu fused at pack
    pack_a_f16<<<ew_blocks((ll)V * KP_GF), TB, 0, stream>>>(
        hv, GF, nullptr, nullptr, 0, nullptr, hv_p, V, KP_GF, 0);
    pack_wt_f16<<<ew_blocks((ll)3 * GF * KP_GF), TB, 0, stream>>>(g1_wih, wih_t, GF, 3 * GF, KP_GF);
    pack_wt_f16<<<ew_blocks((ll)3 * GF * KP_GF), TB, 0, stream>>>(g1_whh, whh_t, GF, 3 * GF, KP_GF);
    gemm_wmma_f16<<<gemm_blocks(V, 3 * GF), TB, 0, stream>>>(cx_p, KP_GF, wih_t, g1_bih, gi, V, 3 * GF, 0);
    gemm_wmma_f16<<<gemm_blocks(V, 3 * GF), TB, 0, stream>>>(hv_p, KP_GF, whh_t, g1_bhh, gh, V, 3 * GF, 0);
    gru_combine_kernel<<<ew_blocks((ll)V * GF), TB, 0, stream>>>(gi, gh, hv, hbuf, V, GF);

    // ---------------- Layer 2: GNNLayer ----------------
    // logits2 = leaky(concat(h[dst], h[src]) @ lpe_w + lpe_b); softmax over dst
    edge_logits_kernel<<<ew_blocks(E), TB, 0, stream>>>(hbuf, dst, hbuf, src, lpe_w, lpe_b, logits, E, GF);
    hipMemsetAsync(menc, 0, (size_t)V * sizeof(unsigned), stream);
    hipMemsetAsync(ssum, 0, (size_t)V * sizeof(float), stream);
    hipMemsetAsync(cbuf, 0, (size_t)V * GF * sizeof(float), stream);
    seg_max_kernel<<<ew_blocks(E), TB, 0, stream>>>(logits, dst, menc, E);
    exp_sum_kernel<<<ew_blocks(E), TB, 0, stream>>>(logits, dst, menc, wexp, ssum, E);
    // hv_proj = h @ lpn_w + lpn_b  (into hv buffer)
    pack_a_f16<<<ew_blocks((ll)V * KP_GF), TB, 0, stream>>>(
        hbuf, GF, nullptr, nullptr, 0, nullptr, hv_p, V, KP_GF, 0);   // h packed
    gemm_wmma_f16<<<gemm_blocks(V, GF), TB, 0, stream>>>(hv_p, KP_GF, lpn_wt, lpn_b, hv, V, GF, 0);
    // c2 = segment_sum(hv_proj[src] * a2, dst)
    attn_gather_scatter<<<ew_blocks((ll)E * GF), TB, 0, stream>>>(hv, wexp, ssum, src, dst, cbuf, E, GF);
    // GRU2 -> output
    pack_a_f16<<<ew_blocks((ll)V * KP_GF), TB, 0, stream>>>(
        cbuf, GF, nullptr, nullptr, 0, nullptr, cx_p, V, KP_GF, 2);   // elu fused at pack
    pack_wt_f16<<<ew_blocks((ll)3 * GF * KP_GF), TB, 0, stream>>>(g2_wih, wih_t, GF, 3 * GF, KP_GF);
    pack_wt_f16<<<ew_blocks((ll)3 * GF * KP_GF), TB, 0, stream>>>(g2_whh, whh_t, GF, 3 * GF, KP_GF);
    gemm_wmma_f16<<<gemm_blocks(V, 3 * GF), TB, 0, stream>>>(cx_p, KP_GF, wih_t, g2_bih, gi, V, 3 * GF, 0);
    gemm_wmma_f16<<<gemm_blocks(V, 3 * GF), TB, 0, stream>>>(hv_p, KP_GF, whh_t, g2_bhh, gh, V, 3 * GF, 0);
    gru_combine_kernel<<<ew_blocks((ll)V * GF), TB, 0, stream>>>(gi, gh, hbuf, (float*)d_out, V, GF);
}
